// SelfAttentionLayer_55009941127664
// MI455X (gfx1250) — matrix operand order repaired and checked by reading, not compiled, not run on hardware
//
#include <hip/hip_runtime.h>

typedef __attribute__((ext_vector_type(16))) _Float16 v16h;
typedef __attribute__((ext_vector_type(8)))  _Float16 v8h;
typedef __attribute__((ext_vector_type(8)))  float    v8f;
typedef __attribute__((ext_vector_type(4)))  float    v4f;

#define DD 512
#define SQ 2048
#define BB 8

// A-fragment (16x32 f16): lane = row (lane%16); contiguous run of 16 halfs at
// (lane/16)*16.  base points at the run.
static __device__ __forceinline__ v16h load_a_f16(const _Float16* base) {
  v8h lo = *(const v8h*)(base);
  v8h hi = *(const v8h*)(base + 8);
  v16h a;
#pragma unroll
  for (int i = 0; i < 8; ++i) { a[i] = lo[i]; a[i + 8] = hi[i]; }
  return a;
}

// Build an A-fragment from fp32 memory (one contiguous 16-float run).
static __device__ __forceinline__ v16h load_a_f32(const float* p) {
  v4f a0 = *(const v4f*)(p);
  v4f a1 = *(const v4f*)(p + 4);
  v4f a2 = *(const v4f*)(p + 8);
  v4f a3 = *(const v4f*)(p + 12);
  v16h a;
#pragma unroll
  for (int i = 0; i < 4; ++i) {
    a[i]      = (_Float16)a0[i];
    a[4 + i]  = (_Float16)a1[i];
    a[8 + i]  = (_Float16)a2[i];
    a[12 + i] = (_Float16)a3[i];
  }
  return a;
}

// Build a B-fragment from fp32 memory (one contiguous 16-float run).
static __device__ __forceinline__ v16h load_b_f32(const float* p) {
  v4f b0 = *(const v4f*)(p);
  v4f b1 = *(const v4f*)(p + 4);
  v4f b2 = *(const v4f*)(p + 8);
  v4f b3 = *(const v4f*)(p + 12);
  v16h b;
#pragma unroll
  for (int i = 0; i < 4; ++i) {
    b[i]      = (_Float16)b0[i];
    b[4 + i]  = (_Float16)b1[i];
    b[8 + i]  = (_Float16)b2[i];
    b[12 + i] = (_Float16)b3[i];
  }
  return b;
}

// ---------------------------------------------------------------------------
// Kernel 1: fused QKV projection, 64x64 output macro-tile per wave.
// y = x @ W^T + b, emitted f16.  Q is pre-scaled by (1/sqrt(D))*log2(e).
// V is stored transposed [B, D, S].
// ---------------------------------------------------------------------------
__global__ __launch_bounds__(256)
void qkv_proj_kernel(const float* __restrict__ x,
                     const float* __restrict__ Wq, const float* __restrict__ bq,
                     const float* __restrict__ Wk, const float* __restrict__ bk,
                     const float* __restrict__ Wv, const float* __restrict__ bv,
                     _Float16* __restrict__ Qh, _Float16* __restrict__ Kh,
                     _Float16* __restrict__ Vth)
{
  const int wave = threadIdx.x >> 5;
  const int lane = threadIdx.x & 31;
  const int r    = lane & 15;
  const int hi   = lane >> 4;
  const int hi8  = hi * 8;
  const int hi16 = hi * 16;

  // 3 matrices * 256 row-groups * 8 col-groups = 6144 wave tiles
  const unsigned gid = blockIdx.x * 8u + (unsigned)wave;
  const int m_idx    = (int)(gid / 2048u);        // 0=Q 1=K 2=V
  const unsigned t   = gid % 2048u;
  const int i0       = (int)(t >> 3) * 64;        // output rows
  const int e0       = (int)(t & 7u) * 64;        // output cols

  const float* W    = (m_idx == 0) ? Wq : (m_idx == 1) ? Wk : Wv;
  const float* bias = (m_idx == 0) ? bq : (m_idx == 1) ? bk : bv;

  const float* xrow[4];
  const float* wrow[4];
#pragma unroll
  for (int a = 0; a < 4; ++a)
    xrow[a] = x + (size_t)(i0 + a * 16 + r) * DD;
#pragma unroll
  for (int c = 0; c < 4; ++c)
    wrow[c] = W + (size_t)(e0 + c * 16 + r) * DD;

  v8f acc[4][4] = {};

  for (int kk = 0; kk < 16; ++kk) {
    const int k0 = kk * 32;
    v16h af[4], bf[4];
#pragma unroll
    for (int a = 0; a < 4; ++a) af[a] = load_a_f32(xrow[a] + k0 + hi16);
#pragma unroll
    for (int c = 0; c < 4; ++c) bf[c] = load_b_f32(wrow[c] + k0 + hi16);
#pragma unroll
    for (int a = 0; a < 4; ++a)
#pragma unroll
      for (int c = 0; c < 4; ++c)
        acc[a][c] = __builtin_amdgcn_wmma_f32_16x16x32_f16(
            false, af[a], false, bf[c], (short)0, acc[a][c], false, false);
  }

  float bias_v[4];
#pragma unroll
  for (int c = 0; c < 4; ++c) bias_v[c] = bias[e0 + c * 16 + r];

  if (m_idx == 0) {
    const float qs = 0.04419417382415922f * 1.4426950408889634f; // scale*log2e
#pragma unroll
    for (int a = 0; a < 4; ++a)
#pragma unroll
      for (int c = 0; c < 4; ++c)
#pragma unroll
        for (int vr = 0; vr < 8; ++vr) {
          const int row = i0 + a * 16 + vr + 8 * hi;
          const int col = e0 + c * 16 + r;
          Qh[(size_t)row * DD + col] = (_Float16)((acc[a][c][vr] + bias_v[c]) * qs);
        }
  } else if (m_idx == 1) {
#pragma unroll
    for (int a = 0; a < 4; ++a)
#pragma unroll
      for (int c = 0; c < 4; ++c)
#pragma unroll
        for (int vr = 0; vr < 8; ++vr) {
          const int row = i0 + a * 16 + vr + 8 * hi;
          const int col = e0 + c * 16 + r;
          Kh[(size_t)row * DD + col] = (_Float16)(acc[a][c][vr] + bias_v[c]);
        }
  } else {
#pragma unroll
    for (int a = 0; a < 4; ++a)
#pragma unroll
      for (int c = 0; c < 4; ++c)
#pragma unroll
        for (int vr = 0; vr < 8; ++vr) {
          const int row = i0 + a * 16 + vr + 8 * hi;  // global token index
          const int bb  = row >> 11;
          const int s   = row & 2047;
          const int col = e0 + c * 16 + r;
          Vth[((size_t)bb * DD + col) * SQ + s] = (_Float16)(acc[a][c][vr] + bias_v[c]);
        }
  }
}

// ---------------------------------------------------------------------------
// Kernel 2: flash attention.  32 queries / block, 8 waves, 64-key tiles.
// Wave w: row-half rh = w>>2 (16 rows), QK column-tile nt = w&3,
// PV output slice = 128 cols at (w&3)*128.
// ---------------------------------------------------------------------------
#define SS2 68   // S_lds row stride (f32), 64 + 4 pad
#define PS2 72   // P_lds row stride (f16), 64 + 8 pad (keeps 16B alignment)

__global__ __launch_bounds__(256)
void flash_attn_kernel(const _Float16* __restrict__ Qh,
                       const _Float16* __restrict__ Kh,
                       const _Float16* __restrict__ Vth,
                       float* __restrict__ out)
{
  __shared__ __align__(16) float    S_lds[32 * SS2];
  __shared__ __align__(16) _Float16 P_lds[32 * PS2];
  __shared__ float rf_lds[32];
  __shared__ float li_lds[32];

  const int wave = threadIdx.x >> 5;
  const int lane = threadIdx.x & 31;
  const int r    = lane & 15;
  const int hi   = lane >> 4;
  const int hi8  = hi * 8;
  const int hi16 = hi * 16;
  const int rh   = wave >> 2;          // row-half: rows [rh*16, rh*16+16)
  const int nt   = wave & 3;           // QK column tile / PV column slice

  const int b  = blockIdx.x >> 6;          // batch
  const int q0 = (blockIdx.x & 63) << 5;   // first query row (32 per block)

  // resident Q fragments for this wave's 16-row half
  const _Float16* Qrow = Qh + (size_t)(b * SQ + q0 + rh * 16 + r) * DD;
  v16h qf[16];
#pragma unroll
  for (int kk = 0; kk < 16; ++kk)
    qf[kk] = load_a_f16(Qrow + kk * 32 + hi16);

  v8f o[8] = {};                        // 16 rows x 128 cols accumulator
  float m_i = -3.0e38f;
  float l_i = 0.0f;

  const int key_lane = nt * 16 + r;     // key column within the 64-key tile
  const int myrow    = 4 * wave + (lane >> 3);  // softmax row of this lane
  const int c0       = lane & 7;                // softmax col phase

  for (int n0 = 0; n0 < SQ; n0 += 64) {
    // ---- S = Q K^T: this wave's 16x16 sub-tile ------------------------
    const _Float16* Krow = Kh + (size_t)(b * SQ + n0 + key_lane) * DD;
    v8f sacc = {};
#pragma unroll
    for (int kk = 0; kk < 16; ++kk) {
      v16h bf = *(const v16h*)(Krow + kk * 32 + hi16);
      sacc = __builtin_amdgcn_wmma_f32_16x16x32_f16(false, qf[kk], false, bf,
                                                    (short)0, sacc, false, false);
    }
#pragma unroll
    for (int vr = 0; vr < 8; ++vr)
      S_lds[(rh * 16 + vr + 8 * hi) * SS2 + nt * 16 + r] = sacc[vr];
    __syncthreads();

    // ---- online softmax: 4 rows per wave, 8 lanes per row --------------
    float v[8];
    float tmax = -3.0e38f;
#pragma unroll
    for (int i = 0; i < 8; ++i) {
      v[i] = S_lds[myrow * SS2 + c0 + 8 * i];
      tmax = fmaxf(tmax, v[i]);
    }
#pragma unroll
    for (int off = 1; off < 8; off <<= 1)
      tmax = fmaxf(tmax, __shfl_xor(tmax, off, 32));
    const float mnew   = fmaxf(m_i, tmax);
    const float rscale = exp2f(m_i - mnew);
    float psum = 0.0f;
#pragma unroll
    for (int i = 0; i < 8; ++i) {
      const float p = exp2f(v[i] - mnew);
      psum += p;
      P_lds[myrow * PS2 + c0 + 8 * i] = (_Float16)p;
    }
#pragma unroll
    for (int off = 1; off < 8; off <<= 1)
      psum += __shfl_xor(psum, off, 32);
    l_i = l_i * rscale + psum;
    m_i = mnew;
    if (c0 == 0) rf_lds[myrow] = rscale;
    __syncthreads();

    // ---- O *= rscale;  O += P @ V --------------------------------------
    float rr[8];
#pragma unroll
    for (int vr = 0; vr < 8; ++vr) rr[vr] = rf_lds[rh * 16 + vr + 8 * hi];
#pragma unroll
    for (int j = 0; j < 8; ++j)
#pragma unroll
      for (int vr = 0; vr < 8; ++vr) o[j][vr] *= rr[vr];

    const _Float16* Vbase = Vth + (size_t)b * DD * SQ + n0;
#pragma unroll
    for (int kk = 0; kk < 2; ++kk) {
      const _Float16* pp = &P_lds[(rh * 16 + r) * PS2 + kk * 32 + hi16];
      v8h plo = *(const v8h*)(pp);
      v8h phi = *(const v8h*)(pp + 8);
      v16h pf;
#pragma unroll
      for (int i = 0; i < 8; ++i) { pf[i] = plo[i]; pf[i + 8] = phi[i]; }
      const int kb = kk * 32 + hi16;
#pragma unroll
      for (int j = 0; j < 8; ++j) {
        const int col = nt * 128 + j * 16 + r;
        v16h vf = *(const v16h*)(Vbase + (size_t)col * SQ + kb);
        o[j] = __builtin_amdgcn_wmma_f32_16x16x32_f16(false, pf, false, vf,
                                                      (short)0, o[j], false, false);
      }
    }
    __syncthreads();
  }

  // ---- finalize: O /= l, write fp32 output ------------------------------
  if (c0 == 0) li_lds[myrow] = 1.0f / l_i;
  __syncthreads();
  float rr[8];
#pragma unroll
  for (int vr = 0; vr < 8; ++vr) rr[vr] = li_lds[rh * 16 + vr + 8 * hi];

  float* orow = out + (size_t)(b * SQ + q0 + rh * 16) * DD;
#pragma unroll
  for (int j = 0; j < 8; ++j) {
    const int col = nt * 128 + j * 16 + r;
#pragma unroll
    for (int vr = 0; vr < 8; ++vr)
      orow[(size_t)(vr + 8 * hi) * DD + col] = o[j][vr] * rr[vr];
  }
}

// ---------------------------------------------------------------------------
extern "C" void kernel_launch(void* const* d_in, const int* in_sizes, int n_in,
                              void* d_out, int out_size, void* d_ws, size_t ws_size,
                              hipStream_t stream) {
  (void)in_sizes; (void)n_in; (void)out_size; (void)ws_size;
  const float* x  = (const float*)d_in[0];
  const float* Wq = (const float*)d_in[1];
  const float* bq = (const float*)d_in[2];
  const float* Wk = (const float*)d_in[3];
  const float* bk = (const float*)d_in[4];
  const float* Wv = (const float*)d_in[5];
  const float* bv = (const float*)d_in[6];
  float* out = (float*)d_out;

  const size_t mat_bytes = (size_t)BB * SQ * DD * sizeof(_Float16); // 16 MB
  char* ws = (char*)d_ws;
  _Float16* Qh  = (_Float16*)(ws);
  _Float16* Kh  = (_Float16*)(ws + mat_bytes);
  _Float16* Vth = (_Float16*)(ws + 2 * mat_bytes);

  // 3 matrices * 256 row-groups * 8 col-groups = 6144 wave tiles / 8 waves
  qkv_proj_kernel<<<768, 256, 0, stream>>>(x, Wq, bq, Wk, bk, Wv, bv,
                                           Qh, Kh, Vth);
  // B * S/32 = 512 query-tile blocks
  flash_attn_kernel<<<512, 256, 0, stream>>>(Qh, Kh, Vth, out);
}
